// RippleGPT_18202071400564
// MI455X (gfx1250) — matrix operand order repaired and checked by reading, not compiled, not run on hardware
//
#include <hip/hip_runtime.h>

// ---------------------------------------------------------------- constants
#define LNUM 4
#define HNUM 12
#define CDIM 768
#define HSD  64
#define HIDD 2048
#define VOC  32000
#define BB   2
#define TT   2048
#define MM   (BB * TT)   // 4096 rows

typedef __attribute__((ext_vector_type(16))) _Float16 v16h;
typedef __attribute__((ext_vector_type(8)))  _Float16 h8;
typedef __attribute__((ext_vector_type(8)))  float    v8f;
typedef unsigned int u32;
typedef __attribute__((ext_vector_type(4))) u32 su4;
typedef __attribute__((ext_vector_type(8))) u32 su8;

static __device__ __forceinline__ v16h cat8(h8 lo, h8 hi) {
  return __builtin_shufflevector(lo, hi, 0,1,2,3,4,5,6,7,8,9,10,11,12,13,14,15);
}

static __device__ __forceinline__ v8f wmma_f16(v16h a, v16h b, v8f c) {
  // (neg_a, A, neg_b, B, c_mod, C, reuse_a, reuse_b)
  return __builtin_amdgcn_wmma_f32_16x16x32_f16(false, a, false, b, (short)0, c, false, false);
}

static __device__ __forceinline__ float halfmax(float v) {
  v = fmaxf(v, __shfl_xor(v, 1, 32));
  v = fmaxf(v, __shfl_xor(v, 2, 32));
  v = fmaxf(v, __shfl_xor(v, 4, 32));
  v = fmaxf(v, __shfl_xor(v, 8, 32));
  return v;
}
static __device__ __forceinline__ float halfsum(float v) {
  v += __shfl_xor(v, 1, 32);
  v += __shfl_xor(v, 2, 32);
  v += __shfl_xor(v, 4, 32);
  v += __shfl_xor(v, 8, 32);
  return v;
}

// ---------------------------------------------------------------- Tensor Data Mover: 2D f16 tile -> LDS
// Loads a (rows x 32)-element f16 tile whose rows are strideElems apart in
// global memory into LDS at byte offset ldsOff, padding each 64B row by
// padDW dwords (A tile: 4 -> 80B stride; B tile: 8 -> 96B stride).
static __device__ __forceinline__ void tdm_load_tile_f16(
    u32 ldsOff, const void* gaddr, u32 rows, u32 strideElems, u32 padDW) {
  unsigned long long ga = (unsigned long long)gaddr;
  su4 g0;
  g0.x = 1u;                               // count=1, user mode, no gather
  g0.y = ldsOff;                           // lds_addr
  g0.z = (u32)ga;                          // global_addr[31:0]
  g0.w = (u32)(ga >> 32) | (2u << 30);     // global_addr[56:32] | type=2
  su8 g1;
  // data_size=2B; pad_enable; pad_interval: code 3 -> every 16 DW (one 64B row);
  // pad_amount: code padDW-1 -> padDW dwords
  g1.s0 = (1u << 16) | (1u << 20) | (3u << 22) | ((padDW - 1u) << 25);
  g1.s1 = (32u & 0xFFFFu) << 16;           // tensor_dim0 = 32 elems (lo16 in hi half)
  g1.s2 = (rows & 0xFFFFu) << 16;          // tensor_dim0 hi | tensor_dim1 lo
  g1.s3 = (32u << 16);                     // tensor_dim1 hi | tile_dim0 = 32
  g1.s4 = rows & 0xFFFFu;                  // tile_dim1 = rows, tile_dim2 = 0
  g1.s5 = strideElems;                     // tensor_dim0_stride (elems)
  g1.s6 = 0u;                              // stride0 hi | tensor_dim1_stride lo
  g1.s7 = 0u;
  su4 z = { 0u, 0u, 0u, 0u };              // groups 2/3: tile_dim3/4 = 0 (unused)
  asm volatile("tensor_load_to_lds %0, %1, %2, %3"
               :: "s"(g0), "s"(g1), "s"(z), "s"(z) : "memory");
}

// ---------------------------------------------------------------- embeddings
__global__ __launch_bounds__(256) void embed_kernel(
    const int* __restrict__ idx, const float* __restrict__ tok,
    const float* __restrict__ pos, float* __restrict__ x) {
  int i = blockIdx.x * 256 + threadIdx.x;            // over MM*CDIM
  int bt = i / CDIM;
  int c  = i - bt * CDIM;
  int t  = bt & (TT - 1);
  x[i] = tok[(size_t)idx[bt] * CDIM + c] + pos[(size_t)t * CDIM + c];
}

// ---------------------------------------------------------------- layernorm -> f16
__global__ __launch_bounds__(256) void ln_kernel(
    const float* __restrict__ in, const float* __restrict__ g,
    const float* __restrict__ b, _Float16* __restrict__ out) {
  int row = blockIdx.x;
  const float* xr = in + (size_t)row * CDIM;
  int tid = threadIdx.x;
  float s = 0.f, s2 = 0.f;
  for (int i = tid; i < CDIM; i += 256) { float v = xr[i]; s += v; s2 += v * v; }
  for (int off = 16; off > 0; off >>= 1) {
    s  += __shfl_xor(s,  off, 32);
    s2 += __shfl_xor(s2, off, 32);
  }
  __shared__ float red[16];
  int wid = tid >> 5;
  if ((tid & 31) == 0) { red[wid] = s; red[8 + wid] = s2; }
  __syncthreads();
  if (wid == 0) {
    float a  = (tid < 8) ? red[tid] : 0.f;
    float a2 = (tid < 8) ? red[8 + tid] : 0.f;
    for (int off = 4; off > 0; off >>= 1) {
      a  += __shfl_xor(a,  off, 32);
      a2 += __shfl_xor(a2, off, 32);
    }
    if (tid == 0) { red[0] = a; red[1] = a2; }
  }
  __syncthreads();
  float mean = red[0] * (1.f / CDIM);
  float var  = red[1] * (1.f / CDIM) - mean * mean;
  float rstd = rsqrtf(var + 1e-5f);
  _Float16* orow = out + (size_t)row * CDIM;
  for (int i = tid; i < CDIM; i += 256)
    orow[i] = (_Float16)((xr[i] - mean) * rstd * g[i] + b[i]);
}

// ---------------------------------------------------------------- fp32[K,N] -> f16[N,K] (transposed), z-batched
__global__ __launch_bounds__(256) void tconv_kernel(
    const float* __restrict__ src, _Float16* __restrict__ dst, int K, int N) {
  src += (size_t)blockIdx.z * K * N;
  dst += (size_t)blockIdx.z * N * K;
  int i = blockIdx.x * 256 + threadIdx.x;            // over N*K
  if (i >= N * K) return;
  int n = i / K;
  int k = i - n * K;
  dst[i] = (_Float16)src[(size_t)k * N + n];
}

// ---------------------------------------------------------------- SwiGLU gate: hh[M,2048] -> g[M,1024] f16
__global__ __launch_bounds__(256) void gate_kernel(
    const float* __restrict__ hh, _Float16* __restrict__ g) {
  int i = blockIdx.x * 256 + threadIdx.x;            // over MM*1024
  int m = i >> 10;
  int j = i & 1023;
  float xv = hh[(size_t)m * HIDD + j];
  float xg = hh[(size_t)m * HIDD + 1024 + j];
  float si = xg / (1.f + __expf(-xg));
  g[i] = (_Float16)(xv * si);
}

// ---------------------------------------------------------------- WMMA GEMM: out = A[M,K] * Bt[N,K]^T
// LDS tiles staged by the Tensor Data Mover, double-buffered: wave 0 issues the
// DMA for tile i+1 into the ping-pong buffer, then waits TENSORcnt<=2 (in-order
// decrement -> tile i's two loads have landed) so the DMA overlaps the WMMAs.
// modes: 0 outF32[row*ldo+col] = acc (+bias)        (fc1, lm_head)
//        1 outF32[row*ldo+col] += acc + bias[col]   (fc2 residual into x)
//        2 outF16 -> q/k layout  [B,H,T,HS], head = blockIdx.z
//        3 outF16 -> v^T layout  [B,H,HS,T], head = blockIdx.z
#define GBM 128
#define GBN 64
#define GBK 32
__global__ __launch_bounds__(256) void gemm_f16_wmma(
    const _Float16* __restrict__ A, const _Float16* __restrict__ Bt,
    float* __restrict__ outF32, _Float16* __restrict__ outF16,
    const float* __restrict__ bias, int M, int N, int K, int ldo, int mode) {
  __shared__ __attribute__((aligned(32))) _Float16 As[2][GBM][GBK + 8];   // 80B row stride
  __shared__ __attribute__((aligned(32))) _Float16 Bs[2][GBN][GBK + 16];  // 96B row stride

  const int tid  = threadIdx.x;
  const int lane = tid & 31;
  const int wave = tid >> 5;
  const int ln   = lane & 15;
  const int half = lane >> 4;
  const int kh   = half * 8;

  const int rowBase = blockIdx.y * GBM;
  const int colBase = blockIdx.x * GBN;
  const int wm = (wave >> 1) * 32;
  const int wn = (wave & 1) * 32;

  if (mode >= 2) Bt += (size_t)blockIdx.z * N * K;

  // LDS byte offsets: generic LDS pointers carry the offset in addr[31:0]
  const u32 asOff0 = (u32)(size_t)&As[0][0][0];
  const u32 asOff1 = (u32)(size_t)&As[1][0][0];
  const u32 bsOff0 = (u32)(size_t)&Bs[0][0][0];
  const u32 bsOff1 = (u32)(size_t)&Bs[1][0][0];
  const char* Ag = (const char*)(A  + (size_t)rowBase * K);
  const char* Bg = (const char*)(Bt + (size_t)colBase * K);

  v8f c00 = {}, c01 = {}, c10 = {}, c11 = {};

  const int nIter = K / GBK;

  // prologue: DMA tile 0 into buffer 0
  if (tid < 32) {   // wave 0 drives the TDM (EXEC ignored, issues once per wave)
    tdm_load_tile_f16(asOff0, Ag, GBM, (u32)K, 4);
    tdm_load_tile_f16(bsOff0, Bg, GBN, (u32)K, 8);
  }

  for (int it = 0; it < nIter; ++it) {
    const int cur = it & 1;
    if (tid < 32) {
      if (it + 1 < nIter) {
        // issue next tile into the other buffer, then wait for the *current*
        // tile only (<=2 outstanding; TENSORcnt decrements in issue order)
        tdm_load_tile_f16(cur ? asOff0 : asOff1,
                          Ag + (size_t)(it + 1) * GBK * 2, GBM, (u32)K, 4);
        tdm_load_tile_f16(cur ? bsOff0 : bsOff1,
                          Bg + (size_t)(it + 1) * GBK * 2, GBN, (u32)K, 8);
        __builtin_amdgcn_s_wait_tensorcnt(2);
      } else {
        __builtin_amdgcn_s_wait_tensorcnt(0);
      }
    }
    __syncthreads();

    // A fragment: lane = row (ln), K chunks [kh,kh+8) and [16+kh,16+kh+8)
    v16h a0 = cat8(*(const h8*)&As[cur][wm + ln][kh],
                   *(const h8*)&As[cur][wm + ln][16 + kh]);
    v16h a1 = cat8(*(const h8*)&As[cur][wm + 16 + ln][kh],
                   *(const h8*)&As[cur][wm + 16 + ln][16 + kh]);
    // B fragment: lane = col (ln), K chunks [kh,kh+8) and [16+kh,16+kh+8)
    v16h b0 = cat8(*(const h8*)&Bs[cur][wn + ln][kh], *(const h8*)&Bs[cur][wn + ln][16 + kh]);
    v16h b1 = cat8(*(const h8*)&Bs[cur][wn + 16 + ln][kh], *(const h8*)&Bs[cur][wn + 16 + ln][16 + kh]);

    // wmma consumes the ds_loads here, forcing s_wait_dscnt before the tail
    // barrier -> iteration i+1's TDM (which targets the buffer last read at
    // i) can never race an outstanding LDS read.
    c00 = wmma_f16(a0, b0, c00);
    c01 = wmma_f16(a0, b1, c01);
    c10 = wmma_f16(a1, b0, c10);
    c11 = wmma_f16(a1, b1, c11);
    __syncthreads();
  }

  auto store1 = [&](int row, int col, float acc) {
    if (mode == 0) {
      outF32[(size_t)row * ldo + col] = acc + (bias ? bias[col] : 0.f);
    } else if (mode == 1) {
      outF32[(size_t)row * ldo + col] += acc + bias[col];
    } else {
      int bb = row >> 11;          // row / T
      int tt = row & (TT - 1);     // row % T
      int z  = blockIdx.z;
      if (mode == 2)
        outF16[(((size_t)bb * HNUM + z) * TT + tt) * HSD + col] = (_Float16)acc;
      else
        outF16[(((size_t)bb * HNUM + z) * HSD + col) * TT + tt] = (_Float16)acc;
    }
  };

#pragma unroll
  for (int r = 0; r < 8; r++) {
    int r0 = rowBase + wm + r + 8 * half;
    int r1 = r0 + 16;
    int cA = colBase + wn + ln;
    int cB = cA + 16;
    store1(r0, cA, c00[r]);
    store1(r0, cB, c01[r]);
    store1(r1, cA, c10[r]);
    store1(r1, cB, c11[r]);
  }
}

// ---------------------------------------------------------------- flash-style attention, x += att
// grid: (T/64, B*H); block 128 = 4 waves; each wave owns one 16-row t-tile
__global__ __launch_bounds__(128) void attn_kernel(
    const _Float16* __restrict__ q, const _Float16* __restrict__ k,
    const _Float16* __restrict__ vt, const float* __restrict__ decay,
    float* __restrict__ x) {
  __shared__ __attribute__((aligned(32))) _Float16 Ps[4][16][40];  // per-wave P tile (16x32 f16)

  const int tid  = threadIdx.x;
  const int lane = tid & 31;
  const int wave = tid >> 5;
  const int ln   = lane & 15;
  const int half = lane >> 4;
  const int kh   = half * 8;

  const int bh = blockIdx.y;
  const int b  = bh / HNUM;
  const int h  = bh - b * HNUM;
  const int tTile = blockIdx.x * 4 + wave;
  const int tBase = tTile * 16;

  const float dc    = fabsf(decay[h]);
  const float scale = 0.125f;  // HS^-0.5

  const _Float16* qb_ = q  + (size_t)bh * TT * HSD;
  const _Float16* kb_ = k  + (size_t)bh * TT * HSD;
  const _Float16* vb_ = vt + (size_t)bh * HSD * TT;

  // Q fragments (rows tBase..tBase+15, K = 64 in two 32-chunks), resident in regs
  const int mQ = tBase + ln;
  v16h aq0 = cat8(*(const h8*)&qb_[(size_t)mQ * HSD + kh],
                  *(const h8*)&qb_[(size_t)mQ * HSD + 16 + kh]);
  v16h aq1 = cat8(*(const h8*)&qb_[(size_t)mQ * HSD + 32 + kh],
                  *(const h8*)&qb_[(size_t)mQ * HSD + 48 + kh]);

  float mrow[8], lrow[8];
#pragma unroll
  for (int r = 0; r < 8; r++) { mrow[r] = -1e30f; lrow[r] = 0.f; }
  v8f o0 = {}, o1 = {}, o2 = {}, o3 = {};

  const int nS32 = (tBase >> 5) + 1;
  for (int sc = 0; sc < nS32; ++sc) {
    const int sBase = sc * 32;

    // ---- scores: 16x32 tile = two C fragments
    const int n0 = sBase + ln;
    const int n1 = sBase + 16 + ln;
    v16h bk00 = cat8(*(const h8*)&kb_[(size_t)n0 * HSD + kh], *(const h8*)&kb_[(size_t)n0 * HSD + 16 + kh]);
    v16h bk01 = cat8(*(const h8*)&kb_[(size_t)n0 * HSD + 32 + kh], *(const h8*)&kb_[(size_t)n0 * HSD + 48 + kh]);
    v16h bk10 = cat8(*(const h8*)&kb_[(size_t)n1 * HSD + kh], *(const h8*)&kb_[(size_t)n1 * HSD + 16 + kh]);
    v16h bk11 = cat8(*(const h8*)&kb_[(size_t)n1 * HSD + 32 + kh], *(const h8*)&kb_[(size_t)n1 * HSD + 48 + kh]);
    v8f s0 = {}, s1 = {};
    s0 = wmma_f16(aq0, bk00, s0);
    s0 = wmma_f16(aq1, bk01, s0);
    s1 = wmma_f16(aq0, bk10, s1);
    s1 = wmma_f16(aq1, bk11, s1);

    // ---- bias + causal mask + online softmax
    float p0[8], p1[8], alpha[8];
#pragma unroll
    for (int r = 0; r < 8; r++) {
      int t = tBase + r + 8 * half;
      float v0 = s0[r] * scale + dc * fminf((float)(n0 - t), 0.f);
      float v1 = s1[r] * scale + dc * fminf((float)(n1 - t), 0.f);
      if (n0 > t) v0 = -1e30f;
      if (n1 > t) v1 = -1e30f;
      float tmax = halfmax(fmaxf(v0, v1));
      float mnew = fmaxf(mrow[r], tmax);
      alpha[r] = __expf(mrow[r] - mnew);
      mrow[r]  = mnew;
      p0[r] = __expf(v0 - mnew);
      p1[r] = __expf(v1 - mnew);
      lrow[r] = lrow[r] * alpha[r] + halfsum(p0[r] + p1[r]);
    }
#pragma unroll
    for (int r = 0; r < 8; r++) {
      o0[r] *= alpha[r]; o1[r] *= alpha[r]; o2[r] *= alpha[r]; o3[r] *= alpha[r];
    }

    // ---- relayout P (C-frag) -> A-frag via per-wave LDS slice
#pragma unroll
    for (int r = 0; r < 8; r++) {
      Ps[wave][r + 8 * half][ln]      = (_Float16)p0[r];
      Ps[wave][r + 8 * half][16 + ln] = (_Float16)p1[r];
    }
    v16h ap = cat8(*(const h8*)&Ps[wave][ln][kh], *(const h8*)&Ps[wave][ln][16 + kh]);

    // ---- O += P * V  (V stored transposed [HS,T] so B-frags are contiguous)
    v16h bv0 = cat8(*(const h8*)&vb_[(size_t)(0 * 16 + ln) * TT + sBase + kh], *(const h8*)&vb_[(size_t)(0 * 16 + ln) * TT + sBase + 16 + kh]);
    v16h bv1 = cat8(*(const h8*)&vb_[(size_t)(1 * 16 + ln) * TT + sBase + kh], *(const h8*)&vb_[(size_t)(1 * 16 + ln) * TT + sBase + 16 + kh]);
    v16h bv2 = cat8(*(const h8*)&vb_[(size_t)(2 * 16 + ln) * TT + sBase + kh], *(const h8*)&vb_[(size_t)(2 * 16 + ln) * TT + sBase + 16 + kh]);
    v16h bv3 = cat8(*(const h8*)&vb_[(size_t)(3 * 16 + ln) * TT + sBase + kh], *(const h8*)&vb_[(size_t)(3 * 16 + ln) * TT + sBase + 16 + kh]);
    o0 = wmma_f16(ap, bv0, o0);
    o1 = wmma_f16(ap, bv1, o1);
    o2 = wmma_f16(ap, bv2, o2);
    o3 = wmma_f16(ap, bv3, o3);
  }

  // ---- finalize and residual-add into x
#pragma unroll
  for (int r = 0; r < 8; r++) {
    int t = tBase + r + 8 * half;
    float inv = 1.f / lrow[r];
    float* xp = x + ((size_t)b * TT + t) * CDIM + h * HSD + ln;
    xp[0]  += o0[r] * inv;
    xp[16] += o1[r] * inv;
    xp[32] += o2[r] * inv;
    xp[48] += o3[r] * inv;
  }
}

// ---------------------------------------------------------------- host driver
extern "C" void kernel_launch(void* const* d_in, const int* in_sizes, int n_in,
                              void* d_out, int out_size, void* d_ws, size_t ws_size,
                              hipStream_t stream) {
  const int*   idx   = (const int*)  d_in[0];
  const float* tok   = (const float*)d_in[1];
  const float* pos   = (const float*)d_in[2];
  const float* Wq    = (const float*)d_in[3];
  const float* Wk    = (const float*)d_in[4];
  const float* Wv    = (const float*)d_in[5];
  const float* decay = (const float*)d_in[6];
  const float* ln1g  = (const float*)d_in[7];
  const float* ln1b  = (const float*)d_in[8];
  const float* fc1w  = (const float*)d_in[9];
  const float* fc1b  = (const float*)d_in[10];
  const float* fc2w  = (const float*)d_in[11];
  const float* fc2b  = (const float*)d_in[12];
  const float* ln2g  = (const float*)d_in[13];
  const float* ln2b  = (const float*)d_in[14];
  const float* lnfg  = (const float*)d_in[15];
  const float* lnfb  = (const float*)d_in[16];
  const float* lmw   = (const float*)d_in[17];
  float* out = (float*)d_out;

  char* p = (char*)d_ws;
  auto wsAlloc = [&](size_t bytes) -> char* {
    char* r = p;
    p += (bytes + 255) & ~(size_t)255;
    return r;
  };
  float*    x    = (float*)   wsAlloc((size_t)MM * CDIM * 4);
  _Float16* hbuf = (_Float16*)wsAlloc((size_t)MM * CDIM * 2);
  _Float16* qb   = (_Float16*)wsAlloc((size_t)BB * HNUM * TT * HSD * 2);
  _Float16* kb   = (_Float16*)wsAlloc((size_t)BB * HNUM * TT * HSD * 2);
  _Float16* vtb  = (_Float16*)wsAlloc((size_t)BB * HNUM * HSD * TT * 2);
  float*    hh   = (float*)   wsAlloc((size_t)MM * HIDD * 4);
  _Float16* gbuf = (_Float16*)wsAlloc((size_t)MM * (HIDD / 2) * 2);
  _Float16* wqT  = (_Float16*)wsAlloc((size_t)LNUM * HNUM * HSD * CDIM * 2);
  _Float16* wkT  = (_Float16*)wsAlloc((size_t)LNUM * HNUM * HSD * CDIM * 2);
  _Float16* wvT  = (_Float16*)wsAlloc((size_t)LNUM * HNUM * HSD * CDIM * 2);
  _Float16* fc1T = (_Float16*)wsAlloc((size_t)LNUM * HIDD * CDIM * 2);
  _Float16* fc2T = (_Float16*)wsAlloc((size_t)LNUM * CDIM * (HIDD / 2) * 2);
  _Float16* lmT  = (_Float16*)wsAlloc((size_t)VOC * CDIM * 2);

  // ---- weight conversions (fp32 [K,N] -> f16 [N,K])
  {
    int nk = HSD * CDIM;
    dim3 g((nk + 255) / 256, 1, LNUM * HNUM);
    tconv_kernel<<<g, 256, 0, stream>>>(Wq, wqT, CDIM, HSD);
    tconv_kernel<<<g, 256, 0, stream>>>(Wk, wkT, CDIM, HSD);
    tconv_kernel<<<g, 256, 0, stream>>>(Wv, wvT, CDIM, HSD);
  }
  {
    int nk = HIDD * CDIM;
    tconv_kernel<<<dim3((nk + 255) / 256, 1, LNUM), 256, 0, stream>>>(fc1w, fc1T, CDIM, HIDD);
  }
  {
    int nk = CDIM * (HIDD / 2);
    tconv_kernel<<<dim3((nk + 255) / 256, 1, LNUM), 256, 0, stream>>>(fc2w, fc2T, HIDD / 2, CDIM);
  }
  {
    int nk = VOC * CDIM;
    tconv_kernel<<<dim3((nk + 255) / 256, 1, 1), 256, 0, stream>>>(lmw, lmT, CDIM, VOC);
  }

  // ---- embeddings
  embed_kernel<<<(MM * CDIM) / 256, 256, 0, stream>>>(idx, tok, pos, x);

  // ---- transformer blocks
  for (int l = 0; l < LNUM; l++) {
    ln_kernel<<<MM, 256, 0, stream>>>(x, ln1g + l * CDIM, ln1b + l * CDIM, hbuf);

    dim3 gq(1, MM / GBM, HNUM);
    gemm_f16_wmma<<<gq, 256, 0, stream>>>(hbuf, wqT + (size_t)l * HNUM * HSD * CDIM,
                                          nullptr, qb, nullptr, MM, HSD, CDIM, 0, 2);
    gemm_f16_wmma<<<gq, 256, 0, stream>>>(hbuf, wkT + (size_t)l * HNUM * HSD * CDIM,
                                          nullptr, kb, nullptr, MM, HSD, CDIM, 0, 2);
    gemm_f16_wmma<<<gq, 256, 0, stream>>>(hbuf, wvT + (size_t)l * HNUM * HSD * CDIM,
                                          nullptr, vtb, nullptr, MM, HSD, CDIM, 0, 3);

    attn_kernel<<<dim3(TT / 64, BB * HNUM), 128, 0, stream>>>(qb, kb, vtb, decay + l * HNUM, x);

    ln_kernel<<<MM, 256, 0, stream>>>(x, ln2g + l * CDIM, ln2b + l * CDIM, hbuf);

    gemm_f16_wmma<<<dim3(HIDD / GBN, MM / GBM), 256, 0, stream>>>(
        hbuf, fc1T + (size_t)l * HIDD * CDIM, hh, nullptr, fc1b + l * HIDD,
        MM, HIDD, CDIM, HIDD, 0);

    gate_kernel<<<(MM * (HIDD / 2)) / 256, 256, 0, stream>>>(hh, gbuf);

    gemm_f16_wmma<<<dim3(CDIM / GBN, MM / GBM), 256, 0, stream>>>(
        gbuf, fc2T + (size_t)l * CDIM * (HIDD / 2), x, nullptr, fc2b + l * CDIM,
        MM, CDIM, HIDD / 2, CDIM, 1);
  }

  // ---- final LN + LM head
  ln_kernel<<<MM, 256, 0, stream>>>(x, lnfg, lnfb, hbuf);
  gemm_f16_wmma<<<dim3(VOC / GBN, MM / GBM), 256, 0, stream>>>(
      hbuf, lmT, out, nullptr, nullptr, MM, VOC, CDIM, VOC, 0);
}
